// RealUpdateLayer2DU_77429670412789
// MI455X (gfx1250) — compile-verified, matrix operand order checked by reading, not run on hardware
//
#include <hip/hip_runtime.h>
#include <hip/hip_bf16.h>
#include <stdint.h>

typedef __attribute__((ext_vector_type(16))) _Float16 v16h;
typedef __attribute__((ext_vector_type(8)))  _Float16 h8;
typedef __attribute__((ext_vector_type(8)))  float    v8f;
typedef __attribute__((ext_vector_type(4)))  float    f4;
typedef __attribute__((ext_vector_type(4)))  int      i4;

#if defined(__has_builtin)
#  if __has_builtin(__builtin_amdgcn_global_load_async_to_lds_b128)
#    define ATH_ASYNC_LDS 1
#  endif
#endif
#ifndef ATH_ASYNC_LDS
#  define ATH_ASYNC_LDS 0
#endif

namespace {

constexpr int B_ = 4, L_ = 256, E_ = 128, H_ = 8, NU_ = 512, NT_ = 256;
constexpr int BL = B_ * L_;        // 1024 query rows
constexpr int CTX_LD = 3 * E_;     // 384

__device__ __forceinline__ v8f wmma_f16(v16h a, v16h b, v8f c) {
  return __builtin_amdgcn_wmma_f32_16x16x32_f16(false, a, false, b, (short)0, c,
                                                false, false);
}

// 16-byte global -> LDS copy; async path when available (ASYNCcnt-tracked).
__device__ __forceinline__ void cp16_g2l(void* lds, const void* g) {
#if ATH_ASYNC_LDS
  __builtin_amdgcn_global_load_async_to_lds_b128((i4*)g, (i4*)lds, 0, 0);
#else
  *(h8*)lds = *(const h8*)g;
#endif
}

__device__ __forceinline__ void cp_join() {
#if ATH_ASYNC_LDS
#  if __has_builtin(__builtin_amdgcn_s_wait_asynccnt)
  __builtin_amdgcn_s_wait_asynccnt(0);
#  else
  asm volatile("s_wait_asynccnt 0x0" ::: "memory");
#  endif
#endif
}

// K index pattern for 16-bit A/B fragments (ISA 7.12.2):
// lanes 0-15 hold K pairs {0..7,16..23}, lanes 16-31 hold {8..15,24..31}
__device__ __forceinline__ int frag_k(int v, int lane) {
  return ((v & 4) << 2) + (((lane >> 4) & 1) << 3) + ((v & 3) << 1);
}

// A fragment: 16x32 (M x K) f16 row-major, given row stride
__device__ __forceinline__ v16h frag_a_ld(const _Float16* base, int stride, int lane) {
  const int r = lane & 15;
  v16h f;
#pragma unroll
  for (int v = 0; v < 8; ++v) {
    int k = frag_k(v, lane);
    f[2 * v]     = base[r * stride + k];
    f[2 * v + 1] = base[r * stride + k + 1];
  }
  return f;
}

// B fragment for B = Xᵀ where X is (N x K) row-major: B[k][n] = X[n][k]
__device__ __forceinline__ v16h frag_bT_ld(const _Float16* base, int stride, int lane) {
  const int n = lane & 15;
  v16h f;
#pragma unroll
  for (int v = 0; v < 8; ++v) {
    int k = frag_k(v, lane);
    f[2 * v]     = base[n * stride + k];
    f[2 * v + 1] = base[n * stride + k + 1];
  }
  return f;
}

// B fragment for B stored (K x N) row-major: B[k][n] = base[k*stride + n]
__device__ __forceinline__ v16h frag_bR_ld(const _Float16* base, int stride, int lane) {
  const int n = lane & 15;
  v16h f;
#pragma unroll
  for (int v = 0; v < 8; ++v) {
    int k = frag_k(v, lane);
    f[2 * v]     = base[k * stride + n];
    f[2 * v + 1] = base[(k + 1) * stride + n];
  }
  return f;
}

// ---------------------------------------------------------------------------
// Tiled GEMM: C[M,N] = A[M,K] @ W[N,K]^T + bias (bias always non-null).
// 64x64 C tile per 256-thread block (8 waves, each 16x32 -> 2 WMMAs/K-step).
// M, N multiples of 64 at every call site.  RAGGED: K not multiple of 32
// and lda not 16B aligned (UU path, K=126) -> clamped scalar loads + selects.
// All hot-path offsets are 32-bit (largest buffer is 2048x256 elements).
// ---------------------------------------------------------------------------
template <typename TI, typename TO, bool RAGGED>
__global__ void __launch_bounds__(256)
gemm_wmma(const TI* __restrict__ A, int lda,
          const TI* __restrict__ W, int ldw,
          const float* __restrict__ bias,
          TO* __restrict__ C, int ldc, int K) {
  __shared__ _Float16 As[64 * 32];
  __shared__ _Float16 Ws[64 * 32];
  const int tid = threadIdx.x, lane = tid & 31, wave = tid >> 5;
  const int wr = (wave >> 1) * 16;   // 0,16,32,48
  const int wc = (wave & 1) * 32;    // 0,32
  const int m0 = blockIdx.x * 64, n0 = blockIdx.y * 64;
  const int sr = tid >> 2;           // staging row 0..63
  const int sc = (tid & 3) * 8;      // staging col {0,8,16,24}
  const int aRow = (m0 + sr) * lda;  // 32-bit row offsets
  const int wRow = (n0 + sr) * ldw;
  v8f acc0 = {}, acc1 = {};

  for (int k0 = 0; k0 < K; k0 += 32) {
    h8 av, wv;
    if constexpr (RAGGED) {
#pragma unroll
      for (int t = 0; t < 8; ++t) {
        int gk = k0 + sc + t;
        int ck = gk < K ? gk : K - 1;            // clamped, always in-bounds
        float a = (float)A[aRow + ck];
        float w = (float)W[wRow + ck];
        av[t] = (gk < K) ? (_Float16)a : (_Float16)0.f;
        wv[t] = (gk < K) ? (_Float16)w : (_Float16)0.f;
      }
    } else if constexpr (sizeof(TI) == 4) {      // fp32 inputs: 2x b128 + cvt
      const float* arow = (const float*)&A[aRow + k0 + sc];
      const float* wrow = (const float*)&W[wRow + k0 + sc];
      f4 a0 = *(const f4*)arow, a1 = *(const f4*)(arow + 4);
      f4 w0 = *(const f4*)wrow, w1 = *(const f4*)(wrow + 4);
#pragma unroll
      for (int t = 0; t < 4; ++t) {
        av[t] = (_Float16)a0[t]; av[t + 4] = (_Float16)a1[t];
        wv[t] = (_Float16)w0[t]; wv[t + 4] = (_Float16)w1[t];
      }
      if (k0 + 32 < K) {                         // near-cache global prefetch
        __builtin_prefetch(arow + 32, 0, 3);
        __builtin_prefetch(wrow + 32, 0, 3);
      }
    } else {                                     // fp16 inputs: 1x b128 each
      av = *(const h8*)&A[aRow + k0 + sc];
      wv = *(const h8*)&W[wRow + k0 + sc];
      if (k0 + 32 < K) {
        __builtin_prefetch(&A[aRow + k0 + 32 + sc], 0, 3);
        __builtin_prefetch(&W[wRow + k0 + 32 + sc], 0, 3);
      }
    }
    *(h8*)&As[sr * 32 + sc] = av;
    *(h8*)&Ws[sr * 32 + sc] = wv;
    __syncthreads();

    v16h af = frag_a_ld(As + wr * 32, 32, lane);
    v16h b0 = frag_bT_ld(Ws + wc * 32, 32, lane);
    v16h b1 = frag_bT_ld(Ws + (wc + 16) * 32, 32, lane);
    acc0 = wmma_f16(af, b0, acc0);
    acc1 = wmma_f16(af, b1, acc1);
    __syncthreads();
  }

  // C/D layout: VGPR j -> (M = j + 8*(lane>>4), N = lane&15)
  const int n = lane & 15, mh = (lane >> 4) << 3;
  const int col0 = n0 + wc + n, col1 = col0 + 16;
  const float bv0 = bias[col0];
  const float bv1 = bias[col1];
#pragma unroll
  for (int j = 0; j < 8; ++j) {
    int m = m0 + wr + j + mh;
    C[m * ldc + col0] = (TO)(acc0[j] + bv0);
    C[m * ldc + col1] = (TO)(acc1[j] + bv1);
  }
}

// ---------------------------------------------------------------------------
// Fold the per-branch output projections into dim_w:
//   W_eff[:, i*E:(i+1)*E] = dim_w[:, i*E:(i+1)*E] @ out_w_i        (fp16)
//   b_eff = dim_b + sum_i dim_w_i @ out_b_i
// ---------------------------------------------------------------------------
__global__ void __launch_bounds__(128)
combine_w(const float* __restrict__ dimw, const float* __restrict__ w0,
          const float* __restrict__ w1, const float* __restrict__ w2,
          _Float16* __restrict__ weff) {
  const int r = blockIdx.x, i = blockIdx.y, c = threadIdx.x;
  const float* wi = (i == 0) ? w0 : (i == 1) ? w1 : w2;
  float acc = 0.f;
  for (int j = 0; j < E_; ++j)
    acc += dimw[r * CTX_LD + i * E_ + j] * wi[j * E_ + c];
  weff[r * CTX_LD + i * E_ + c] = (_Float16)acc;
}

__global__ void __launch_bounds__(128)
beff_k(const float* __restrict__ dimw, const float* __restrict__ dimb,
       const float* __restrict__ b0, const float* __restrict__ b1,
       const float* __restrict__ b2, float* __restrict__ beff) {
  const int r = threadIdx.x;
  float acc = dimb[r];
  for (int j = 0; j < E_; ++j) {
    acc += dimw[r * CTX_LD + j] * b0[j]
         + dimw[r * CTX_LD + E_ + j] * b1[j]
         + dimw[r * CTX_LD + 2 * E_ + j] * b2[j];
  }
  beff[r] = acc;
}

// ---------------------------------------------------------------------------
// Attention: one wave per (batch b, head h, 16-query tile).
// NKEYS keys per batch; D=16 zero-padded to K=32 for WMMA (pads written once).
// Score pass: 32 keys / iteration (2 WMMAs).  K/V staged with async-to-LDS.
// CI path (UU2DU): rank-2 logit correction + rank-1 value correction from
// Wk/Wv columns 126/127.  Output goes into concatenated fp16 ctx[BL, 384].
// ---------------------------------------------------------------------------
template <int NKEYS, bool CI>
__global__ void __launch_bounds__(32)
attn_kernel(const _Float16* __restrict__ q, int ldq,
            const _Float16* __restrict__ kv, int ldkv, int kOfs, int vOfs,
            int keyRowsPerB,
            const float* __restrict__ CIm,   // [B, NU, 2L], CI only
            const float* __restrict__ inw,   // uu_in_w,     CI only
            _Float16* __restrict__ ctx, int slotOfs) {
  __shared__ float    s[16 * NKEYS];     // logits -> exp
  __shared__ _Float16 p[16 * NKEYS];     // normalized probs (A matrix of O pass)
  __shared__ _Float16 qs[16 * 32];       // scaled Q tile (K-padded)
  __shared__ _Float16 kst[32 * 32];      // staged K chunk (K-padded)
  __shared__ _Float16 vst[32 * 16];      // staged V chunk
  __shared__ float alpha[16], beta[16], g0[16], g1[16];

  const int lane = threadIdx.x;
  const int bid = blockIdx.x;
  const int lt = bid & 15, h = (bid >> 4) & 7, bb = bid >> 7;
  const int l0 = lt * 16, rowBase = bb * L_ + l0;

  {  // Q tile (fold 1/sqrt(D)=0.25) + zero all pad regions once
    const int r = lane >> 1, pp = (lane & 1) * 8;
    h8 qv = *(const h8*)&q[(rowBase + r) * ldq + h * 16 + pp];
    h8 qz = {};
#pragma unroll
    for (int t = 0; t < 8; ++t) qv[t] = (_Float16)((float)qv[t] * 0.25f);
    *(h8*)&qs[r * 32 + pp] = qv;
    *(h8*)&qs[r * 32 + 16 + pp] = qz;
    *(h8*)&kst[lane * 32 + 16] = qz;
    *(h8*)&kst[lane * 32 + 24] = qz;
  }
  __syncthreads();

  if (CI && lane < 16) {  // alpha = q.Wk[:,126]_h ; beta = q.Wk[:,127]_h (scaled)
    float a = 0.f, bt = 0.f;
#pragma unroll
    for (int d = 0; d < 16; ++d) {
      float qv = (float)qs[lane * 32 + d];
      a  += qv * inw[(E_ + h * 16 + d) * E_ + 126];
      bt += qv * inw[(E_ + h * 16 + d) * E_ + 127];
    }
    alpha[lane] = a; beta[lane] = bt;
  }
  __syncthreads();

  const v16h af = frag_a_ld(qs, 32, lane);

  // ---- base logits S = Q @ K^T, 32 keys per iteration ----
  for (int kc = 0; kc < NKEYS / 32; ++kc) {
    const int krow = (bb * keyRowsPerB + kc * 32 + lane) * ldkv + kOfs + h * 16;
    cp16_g2l(&kst[lane * 32 + 0], &kv[krow]);
    cp16_g2l(&kst[lane * 32 + 8], &kv[krow + 8]);
    cp_join();
    __syncthreads();
    v8f a0 = {}, a1 = {};
    a0 = wmma_f16(af, frag_bT_ld(kst, 32, lane), a0);
    a1 = wmma_f16(af, frag_bT_ld(kst + 16 * 32, 32, lane), a1);
    const int n = lane & 15, mh = (lane >> 4) << 3;
#pragma unroll
    for (int j = 0; j < 8; ++j) {
      s[(j + mh) * NKEYS + kc * 32 + n] = a0[j];
      s[(j + mh) * NKEYS + kc * 32 + 16 + n] = a1[j];
    }
    __syncthreads();
  }

  // ---- CI logit correction + softmax; 2 lanes per row, shfl_xor combine ----
  {
    const int r = lane & 15, hf = lane >> 4;
    const int nBeg = hf * (NKEYS / 2), nEnd = nBeg + NKEYS / 2;
    float* srow = s + r * NKEYS;
    const float* ciP = CI ? (CIm + bb * NU_ * (2 * L_) + (l0 + r)) : nullptr;
    const float al = CI ? alpha[r] : 0.f;
    const float be = CI ? beta[r]  : 0.f;
    float mx = -3.0e38f;
    for (int n2 = nBeg; n2 < nEnd; ++n2) {
      float v = srow[n2];
      if (CI) v += al * ciP[n2 * (2 * L_)] + be * ciP[n2 * (2 * L_) + L_];
      srow[n2] = v;
      mx = fmaxf(mx, v);
    }
    mx = fmaxf(mx, __shfl_xor(mx, 16, 32));
    float sum = 0.f, gg0 = 0.f, gg1 = 0.f;
    for (int n2 = nBeg; n2 < nEnd; ++n2) {
      float e = __expf(srow[n2] - mx);
      srow[n2] = e;
      sum += e;
      if (CI) {
        gg0 += e * ciP[n2 * (2 * L_)];
        gg1 += e * ciP[n2 * (2 * L_) + L_];
      }
    }
    sum += __shfl_xor(sum, 16, 32);
    if (CI) { gg0 += __shfl_xor(gg0, 16, 32); gg1 += __shfl_xor(gg1, 16, 32); }
    const float rinv = 1.f / sum;
    for (int n2 = nBeg; n2 < nEnd; ++n2)
      p[r * NKEYS + n2] = (_Float16)(srow[n2] * rinv);
    if (CI && hf == 0) { g0[r] = gg0 * rinv; g1[r] = gg1 * rinv; }
  }
  __syncthreads();

  // ---- O = A @ V, 32-key chunks ----
  v8f oc = {};
  for (int kc2 = 0; kc2 < NKEYS / 32; ++kc2) {
    const int vrow = (bb * keyRowsPerB + kc2 * 32 + lane) * ldkv + vOfs + h * 16;
    cp16_g2l(&vst[lane * 16 + 0], &kv[vrow]);
    cp16_g2l(&vst[lane * 16 + 8], &kv[vrow + 8]);
    cp_join();
    __syncthreads();
    v16h pa = frag_a_ld(p + kc2 * 32, NKEYS, lane);
    v16h vb = frag_bR_ld(vst, 16, lane);
    oc = wmma_f16(pa, vb, oc);
    __syncthreads();
  }

  // ---- rank-1 V correction (CI) + store into concatenated ctx ----
  const int n = lane & 15, mh = (lane >> 4) << 3;
  float w0 = 0.f, w1 = 0.f;
  if (CI) {
    w0 = inw[(2 * E_ + h * 16 + n) * E_ + 126];
    w1 = inw[(2 * E_ + h * 16 + n) * E_ + 127];
  }
#pragma unroll
  for (int j = 0; j < 8; ++j) {
    int m = j + mh;
    float v = oc[j];
    if (CI) v += g0[m] * w0 + g1[m] * w1;
    ctx[(rowBase + m) * CTX_LD + slotOfs + h * 16 + n] = (_Float16)v;
  }
}

// ---------------------------------------------------------------------------
// BatchNorm over (B,L) per channel, then normalize + ReLU.
// ---------------------------------------------------------------------------
__global__ void __launch_bounds__(128)
bn_stats(const float* __restrict__ y, float* __restrict__ mv) {
  const int c = threadIdx.x;
  float s = 0.f, s2 = 0.f;
  for (int r = 0; r < BL; ++r) {
    float v = y[r * E_ + c];
    s += v; s2 += v * v;
  }
  float m = s * (1.f / BL);
  mv[c] = m;
  mv[E_ + c] = s2 * (1.f / BL) - m * m;
}

__global__ void __launch_bounds__(256)
bn_apply(const float* __restrict__ y, const float* __restrict__ mv,
         const float* __restrict__ gamma, const float* __restrict__ beta,
         float* __restrict__ out) {
  const int i = blockIdx.x * blockDim.x + threadIdx.x;
  if (i >= BL * E_) return;
  const int c = i & (E_ - 1);
  float v = (y[i] - mv[c]) * rsqrtf(mv[E_ + c] + 1e-5f) * gamma[c] + beta[c];
  out[i] = v > 0.f ? v : 0.f;
}

} // namespace

extern "C" void kernel_launch(void* const* d_in, const int* in_sizes, int n_in,
                              void* d_out, int out_size, void* d_ws, size_t ws_size,
                              hipStream_t stream) {
  (void)in_sizes; (void)n_in; (void)out_size; (void)ws_size;
  const float* UU       = (const float*)d_in[0];   // [B, NU, 126]
  const float* DU       = (const float*)d_in[1];   // [B, L, E]
  const float* TA       = (const float*)d_in[2];   // [B, NT, E]
  const float* CIm      = (const float*)d_in[3];   // [B, NU, 2L]
  const float* uu_in_w  = (const float*)d_in[4];
  const float* uu_in_b  = (const float*)d_in[5];
  const float* uu_out_w = (const float*)d_in[6];
  const float* uu_out_b = (const float*)d_in[7];
  const float* ta_in_w  = (const float*)d_in[8];
  const float* ta_in_b  = (const float*)d_in[9];
  const float* ta_out_w = (const float*)d_in[10];
  const float* ta_out_b = (const float*)d_in[11];
  const float* du_in_w  = (const float*)d_in[12];
  const float* du_in_b  = (const float*)d_in[13];
  const float* du_out_w = (const float*)d_in[14];
  const float* du_out_b = (const float*)d_in[15];
  const float* dim_w    = (const float*)d_in[16];
  const float* dim_b    = (const float*)d_in[17];
  const float* gamma    = (const float*)d_in[18];
  const float* beta     = (const float*)d_in[19];

  char* ws = (char*)d_ws;
  size_t off = 0;
  auto take = [&](size_t bytes) -> char* {
    char* p = ws + off;
    off += (bytes + 255) & ~(size_t)255;
    return p;
  };
  _Float16* ctx    = (_Float16*)take((size_t)BL * CTX_LD * 2);       // [1024,384]
  _Float16* q_uu   = (_Float16*)take((size_t)BL * E_ * 2);           // [1024,128]
  _Float16* kv_uu  = (_Float16*)take((size_t)B_ * NU_ * 2 * E_ * 2); // [2048,256]
  _Float16* q_ta   = (_Float16*)take((size_t)BL * E_ * 2);
  _Float16* kv_ta  = (_Float16*)take((size_t)B_ * NT_ * 2 * E_ * 2); // [1024,256]
  _Float16* qkv_du = (_Float16*)take((size_t)BL * CTX_LD * 2);       // [1024,384]
  _Float16* weff   = (_Float16*)take((size_t)E_ * CTX_LD * 2);       // [128,384]
  float*    beff   = (float*)take((size_t)E_ * 4);
  float*    y      = (float*)take((size_t)BL * E_ * 4);              // [1024,128]
  float*    mv     = (float*)take((size_t)2 * E_ * 4);               // mean|var

  // ---- projections (fp32 -> fp16 WMMA GEMM, 64x64 tiles) ----
  gemm_wmma<float, _Float16, false><<<dim3(BL / 64, E_ / 64), 256, 0, stream>>>(
      DU, E_, uu_in_w, E_, uu_in_b, q_uu, E_, E_);
  gemm_wmma<float, _Float16, true><<<dim3(B_ * NU_ / 64, 2 * E_ / 64), 256, 0, stream>>>(
      UU, E_ - 2, uu_in_w + E_ * E_, E_, uu_in_b + E_, kv_uu, 2 * E_, E_ - 2);
  gemm_wmma<float, _Float16, false><<<dim3(BL / 64, E_ / 64), 256, 0, stream>>>(
      DU, E_, ta_in_w, E_, ta_in_b, q_ta, E_, E_);
  gemm_wmma<float, _Float16, false><<<dim3(B_ * NT_ / 64, 2 * E_ / 64), 256, 0, stream>>>(
      TA, E_, ta_in_w + E_ * E_, E_, ta_in_b + E_, kv_ta, 2 * E_, E_);
  gemm_wmma<float, _Float16, false><<<dim3(BL / 64, CTX_LD / 64), 256, 0, stream>>>(
      DU, E_, du_in_w, E_, du_in_b, qkv_du, CTX_LD, E_);

  // ---- fold output projections into dim_w ----
  combine_w<<<dim3(E_, 3), 128, 0, stream>>>(dim_w, uu_out_w, ta_out_w, du_out_w, weff);
  beff_k<<<1, 128, 0, stream>>>(dim_w, dim_b, uu_out_b, ta_out_b, du_out_b, beff);

  // ---- attention (one wave per (b,h,qtile)); ctx slots: [uu | ta | du] ----
  const int nblk = B_ * H_ * (L_ / 16);  // 512
  attn_kernel<NU_, true><<<nblk, 32, 0, stream>>>(
      q_uu, E_, kv_uu, 2 * E_, 0, E_, NU_, CIm, uu_in_w, ctx, 0);
  attn_kernel<NT_, false><<<nblk, 32, 0, stream>>>(
      q_ta, E_, kv_ta, 2 * E_, 0, E_, NT_, nullptr, nullptr, ctx, E_);
  attn_kernel<NT_, false><<<nblk, 32, 0, stream>>>(
      qkv_du, CTX_LD, qkv_du, CTX_LD, E_, 2 * E_, NT_, nullptr, nullptr, ctx, 2 * E_);

  // ---- y = ctx @ W_eff^T + b_eff  (fp16 x fp16 -> fp32) ----
  gemm_wmma<_Float16, float, false><<<dim3(BL / 64, E_ / 64), 256, 0, stream>>>(
      ctx, CTX_LD, weff, CTX_LD, beff, y, E_, CTX_LD);

  // ---- batchnorm + relu ----
  bn_stats<<<1, 128, 0, stream>>>(y, mv);
  bn_apply<<<(BL * E_ + 255) / 256, 256, 0, stream>>>(y, mv, gamma, beta, (float*)d_out);
}
